// SML_30313879176040
// MI455X (gfx1250) — matrix operand order repaired and checked by reading, not compiled, not run on hardware
//
#include <hip/hip_runtime.h>
#include <cstdint>
#include <cstddef>

// ---------------------------------------------------------------------------
// VQ codebook EMA + soft combine, fused for MI455X (gfx1250, wave32, WMMA).
//   x: [8,512,64,64] f32, m: [1024,512] f32 -> out: [8,512,64,64] f32
// tokens N=32768 (64 per WG => full w-row at fixed b,h), C=512, K=1024 codes.
// Codebook / new_m^T panels staged via GLOBAL_LOAD_ASYNC_TO_LDS (ASYNCcnt),
// double-buffered; WMMA fragment loads software-pipelined.
// ---------------------------------------------------------------------------

#define XN_LD 520    // bf16 token tile row stride   (16B-aligned rows, bank-skewed)
#define BB_LD 520    // bf16 staged codebook tile row stride
#define SF_LD 1040   // bf16 soft matrix row stride (1024 + 16 pad)
#define BL_LD 136    // bf16 new_m^T panel row stride (128 + 8 pad)

typedef __attribute__((ext_vector_type(8)))  __bf16 bf16x8;
typedef __attribute__((ext_vector_type(16))) __bf16 bf16x16;
typedef __attribute__((ext_vector_type(8)))  float  f32x8;

typedef __attribute__((address_space(3))) unsigned char lds_byte;

__device__ __forceinline__ unsigned lds_off(const void* p) {
  return (unsigned)(unsigned long long)(lds_byte*)p;
}
// Async global->LDS 16B copy (no VGPR data; tracked by ASYNCcnt).
__device__ __forceinline__ void async_g2l_b128(const void* g, void* l) {
  unsigned lofs = lds_off(l);
  unsigned long long ga = (unsigned long long)(uintptr_t)g;
  asm volatile("global_load_async_to_lds_b128 %0, %1, off"
               :: "v"(lofs), "v"(ga) : "memory");
}
__device__ __forceinline__ void wait_async0() {
  asm volatile("s_wait_asynccnt 0" ::: "memory");
}

__device__ __forceinline__ unsigned short f2bfu(float f) {
  unsigned u = __float_as_uint(f);                       // RNE float->bf16
  return (unsigned short)((u + 0x7FFFu + ((u >> 16) & 1u)) >> 16);
}
__device__ __forceinline__ float shflx_f(float v, int lane, int mask) {
  return __int_as_float(__builtin_amdgcn_ds_bpermute((lane ^ mask) << 2,
                                                     __float_as_int(v)));
}
__device__ __forceinline__ int shflx_i(int v, int lane, int mask) {
  return __builtin_amdgcn_ds_bpermute((lane ^ mask) << 2, v);
}
__device__ __forceinline__ f32x8 wmma_bf16(bf16x16 a, bf16x16 b, f32x8 c) {
  return __builtin_amdgcn_wmma_f32_16x16x32_bf16(false, a, false, b,
                                                 (short)0, c, false, false);
}
// A fragment (16x32 bf16): lane=row; K = half*8+[0..7] in v0-3, +16.. in v4-7.
__device__ __forceinline__ bf16x16 load_a_frag(const unsigned short* p, int half) {
  const __bf16* q = (const __bf16*)p;
  bf16x8 lo = *(const bf16x8*)(q + half * 8);
  bf16x8 hi = *(const bf16x8*)(q + 16 + half * 8);
  return __builtin_shufflevector(lo, hi, 0,1,2,3,4,5,6,7,8,9,10,11,12,13,14,15);
}
// B fragment (32x16 bf16): lane=col; K = half*8+[0..7] in v0-3, +16.. in v4-7.
__device__ __forceinline__ bf16x16 load_b_frag(const unsigned short* p, int half) {
  const __bf16* q = (const __bf16*)p;
  bf16x8 lo = *(const bf16x8*)(q + half * 8);
  bf16x8 hi = *(const bf16x8*)(q + 16 + half * 8);
  return __builtin_shufflevector(lo, hi, 0,1,2,3,4,5,6,7,8,9,10,11,12,13,14,15);
}

// Async-stage one 16x512 bf16 codebook tile into LDS (8 chunks / thread).
__device__ __forceinline__ void stage_btile(const unsigned short* __restrict__ src,
                                            unsigned short* bb) {
  const int tid = threadIdx.x;
  #pragma unroll
  for (int i = 0; i < 8; ++i) {
    const int e8 = (tid + i * 128) * 8;      // ushort index in 16x512 tile
    async_g2l_b128(src + e8, bb + (e8 >> 9) * BB_LD + (e8 & 511));
  }
}
// Async-stage one 128ch x 128k bf16 new_m^T panel into LDS (16 chunks/thread).
__device__ __forceinline__ void stage_panel(const unsigned short* __restrict__ nmT,
                                            int c0, int k0, unsigned short* bl) {
  const int tid = threadIdx.x;
  #pragma unroll
  for (int i = 0; i < 16; ++i) {
    const int chunk = tid + i * 128;         // 2048 chunks
    const int row = chunk >> 4, col8 = (chunk & 15) * 8;
    async_g2l_b128(nmT + (size_t)(c0 + row) * 1024 + k0 + col8,
                   bl + row * BL_LD + col8);
  }
}

// Stage 64 tokens (full w-row) as l2-normalized bf16 into LDS. 128 threads.
__device__ __forceinline__ void stage_xn(const float* __restrict__ base,
                                         const float* __restrict__ rnx, int tb,
                                         unsigned short* xn) {
  const int tid = threadIdx.x;
  const int w = tid & 63, cp = tid >> 6;     // cp = 0..1
  const float sc = rnx[tb + w];
  for (int c = cp; c < 512; c += 2)
    xn[w * XN_LD + c] = f2bfu(base[(size_t)c * 4096 + w] * sc);
  __syncthreads();
}

// --------------------------- K0: zero scratch ------------------------------
__global__ __launch_bounds__(256) void SML_k_zero(float* __restrict__ esum,
                                                  float* __restrict__ counts) {
  const int t = blockIdx.x * 256 + threadIdx.x;   // grid covers 512*1024
  esum[t] = 0.f;
  if (t < 1024) counts[t] = 0.f;
}

// --------------- K1a: token inverse norms (one (b,h) row / block) ----------
__global__ __launch_bounds__(256) void SML_k_normx(const float* __restrict__ x,
                                                   float* __restrict__ rnx) {
  __shared__ float part[4][64];
  const int tb = blockIdx.x * 64;
  const int b = tb >> 12, hw = tb & 4095;
  const float* base = x + (size_t)b * (512u * 4096u) + hw;
  const int w = threadIdx.x & 63, cp = threadIdx.x >> 6;
  float ss = 0.f;
  for (int c = cp; c < 512; c += 4) { float v = base[(size_t)c * 4096 + w]; ss += v * v; }
  part[cp][w] = ss;
  __syncthreads();
  if (threadIdx.x < 64) {
    const int t = threadIdx.x;
    float s = part[0][t] + part[1][t] + part[2][t] + part[3][t];
    rnx[tb + t] = 1.f / fmaxf(sqrtf(s), 1e-12f);
  }
}

// ------------------- K1b: l2-normalize codebook -> bf16 --------------------
__global__ __launch_bounds__(256) void SML_k_normm(const float* __restrict__ m,
                                                   unsigned short* __restrict__ mnbf) {
  const int row  = blockIdx.x * 8 + (threadIdx.x >> 5);
  const int lane = threadIdx.x & 31;
  const float* p = m + row * 512;
  float ss = 0.f;
  for (int c = lane; c < 512; c += 32) { float v = p[c]; ss += v * v; }
  #pragma unroll
  for (int msk = 1; msk < 32; msk <<= 1) ss += shflx_f(ss, lane, msk);
  const float sc = 1.f / fmaxf(sqrtf(ss), 1e-12f);
  for (int c = lane; c < 512; c += 32) mnbf[row * 512 + c] = f2bfu(p[c] * sc);
}

// --- K2: GEMM1 (xn @ mn^T) + online softmax stats + argmax EMA scatter -----
// WG = 128 thr / 4 waves / 64 tokens. Wave w owns row tile w (16 tokens) and
// sees ALL 1024 codes; codebook tiles async-staged, double-buffered.
__global__ __launch_bounds__(128) void SML_k_score(
    const float* __restrict__ x, const unsigned short* __restrict__ mnbf,
    const float* __restrict__ rnx, float* __restrict__ rowMax,
    float* __restrict__ rowSum, float* __restrict__ counts,
    float* __restrict__ esum) {
  extern __shared__ char smem[];
  unsigned short* xn  = (unsigned short*)smem;                // 64*XN_LD
  unsigned short* bb0 = xn + 64 * XN_LD;                      // 16*BB_LD
  unsigned short* bb1 = bb0 + 16 * BB_LD;                     // 16*BB_LD
  int* fArg = (int*)(bb1 + 16 * BB_LD);                       // 64

  const int tb = blockIdx.x * 64;
  const int b = tb >> 12, hw = tb & 4095;
  const float* base = x + (size_t)b * (512u * 4096u) + hw;
  stage_xn(base, rnx, tb, xn);

  const int tid = threadIdx.x;
  const int wave = tid >> 5, lane = tid & 31, r = lane & 15, half = lane >> 4;
  const unsigned short* arow = xn + (wave * 16 + r) * XN_LD;

  float lmax[8], lsum[8]; int larg[8];
  #pragma unroll
  for (int v = 0; v < 8; ++v) { lmax[v] = -__builtin_inff(); lsum[v] = 0.f; larg[v] = 0; }

  stage_btile(mnbf, bb0);
  wait_async0();
  __syncthreads();
  for (int t = 0; t < 64; ++t) {
    const unsigned short* cur = (t & 1) ? bb1 : bb0;
    unsigned short*       nxt = (t & 1) ? bb0 : bb1;
    if (t < 63) stage_btile(mnbf + (t + 1) * 8192, nxt);   // async, overlaps
    const unsigned short* brow = cur + r * BB_LD;
    f32x8 acc = {};
    bf16x16 a0 = load_a_frag(arow, half);
    bf16x16 b0 = load_b_frag(brow, half);
    #pragma unroll
    for (int k0 = 0; k0 < 480; k0 += 32) {     // pipelined fragment loads
      bf16x16 a1 = load_a_frag(arow + k0 + 32, half);
      bf16x16 b1 = load_b_frag(brow + k0 + 32, half);
      acc = wmma_bf16(a0, b0, acc);
      a0 = a1; b0 = b1;
    }
    acc = wmma_bf16(a0, b0, acc);
    const int col = t * 16 + r;
    #pragma unroll
    for (int v = 0; v < 8; ++v) {              // online softmax + argmax
      float sc = acc[v];
      if (sc > lmax[v]) {
        lsum[v] = lsum[v] * __expf(lmax[v] - sc) + 1.f;
        lmax[v] = sc; larg[v] = col;
      } else {
        lsum[v] += __expf(sc - lmax[v]);
      }
    }
    wait_async0();                             // our async stores to nxt done
    __syncthreads();                           // publish; all done with cur
  }
  // butterfly all-reduce across the 16 lanes of each half (wave32)
  #pragma unroll
  for (int msk = 1; msk < 16; msk <<= 1) {
    #pragma unroll
    for (int v = 0; v < 8; ++v) {
      float om = shflx_f(lmax[v], lane, msk);
      float os = shflx_f(lsum[v], lane, msk);
      int   oa = shflx_i(larg[v], lane, msk);
      float nm = fmaxf(lmax[v], om);
      float ns = lsum[v] * __expf(lmax[v] - nm) + os * __expf(om - nm);
      if (om > lmax[v] || (om == lmax[v] && oa < larg[v])) larg[v] = oa;
      lmax[v] = nm; lsum[v] = ns;
    }
  }
  if (r == 0) {
    #pragma unroll
    for (int v = 0; v < 8; ++v) {
      const int row = wave * 16 + v + 8 * half;
      rowMax[tb + row] = lmax[v];
      rowSum[tb + row] = lsum[v];
      fArg[row] = larg[v];
      atomicAdd(&counts[larg[v]], 1.0f);
    }
  }
  __syncthreads();
  // scatter un-normalized token rows into esum[argmax] (x lines are L0-hot)
  for (int row = 0; row < 64; ++row) {
    float* dst = esum + fArg[row] * 512;
    for (int c = tid; c < 512; c += 128)
      atomicAdd(dst + c, base[(size_t)c * 4096 + row]);
  }
}

// --------- K3: EMA update, written transposed bf16: nmT[c*1024+k] ----------
__global__ __launch_bounds__(256) void SML_k_update(
    const float* __restrict__ m, const float* __restrict__ esum,
    const float* __restrict__ counts, unsigned short* __restrict__ nmT) {
  const int t = blockIdx.x * 256 + threadIdx.x;   // 512*1024 threads
  const int c = t >> 10, k = t & 1023;
  const float cnt = counts[k];
  const float nm  = m[k * 512 + c] * 0.999f +
                    (esum[k * 512 + c] / (cnt + 1e-6f)) * 0.001f;
  nmT[t] = f2bfu(nm);
}

// --- K4: recompute scores -> soft (LDS) -> GEMM2 (soft @ new_m) -> NCHW ----
// Phase 1 fills soft[64x1024] bf16 in LDS (codebook tiles async-pipelined).
// Phase 2: new_m^T panels async double-buffered (aliasing the dead xn/bb
// region); wave w owns row tile w, acc = 8 col tiles.
__global__ __launch_bounds__(128) void SML_k_out(
    const float* __restrict__ x, const unsigned short* __restrict__ mnbf,
    const unsigned short* __restrict__ nmT, const float* __restrict__ rnx,
    const float* __restrict__ rowMax, const float* __restrict__ rowSum,
    float* __restrict__ out) {
  extern __shared__ char smem[];
  unsigned short* xn    = (unsigned short*)smem;              // 64*XN_LD  (phase1)
  unsigned short* bb0   = xn + 64 * XN_LD;                    // 16*BB_LD  (phase1)
  unsigned short* bb1   = bb0 + 16 * BB_LD;                   // 16*BB_LD  (phase1)
  unsigned short* bl0   = (unsigned short*)smem;              // 128*BL_LD (phase2, alias)
  unsigned short* bl1   = bl0 + 128 * BL_LD;                  // 128*BL_LD (phase2, alias)
  unsigned short* softb = bb1 + 16 * BB_LD;                   // 64*SF_LD

  const int tb = blockIdx.x * 64;
  const int b = tb >> 12, hw = tb & 4095;
  const float* base = x + (size_t)b * (512u * 4096u) + hw;
  stage_xn(base, rnx, tb, xn);

  const int tid = threadIdx.x;
  const int wave = tid >> 5, lane = tid & 31, r = lane & 15, half = lane >> 4;
  const unsigned short* arow = xn + (wave * 16 + r) * XN_LD;

  float rm[8], ri[8];
  #pragma unroll
  for (int v = 0; v < 8; ++v) {
    const int row = wave * 16 + v + 8 * half;
    rm[v] = rowMax[tb + row];
    ri[v] = 1.f / rowSum[tb + row];
  }

  // ---- phase 1: soft = softmax(xn @ mn^T) into LDS ----
  stage_btile(mnbf, bb0);
  wait_async0();
  __syncthreads();
  for (int t = 0; t < 64; ++t) {
    const unsigned short* cur = (t & 1) ? bb1 : bb0;
    unsigned short*       nxt = (t & 1) ? bb0 : bb1;
    if (t < 63) stage_btile(mnbf + (t + 1) * 8192, nxt);
    const unsigned short* brow = cur + r * BB_LD;
    f32x8 sacc = {};
    bf16x16 a0 = load_a_frag(arow, half);
    bf16x16 b0 = load_b_frag(brow, half);
    #pragma unroll
    for (int k0 = 0; k0 < 480; k0 += 32) {
      bf16x16 a1 = load_a_frag(arow + k0 + 32, half);
      bf16x16 b1 = load_b_frag(brow + k0 + 32, half);
      sacc = wmma_bf16(a0, b0, sacc);
      a0 = a1; b0 = b1;
    }
    sacc = wmma_bf16(a0, b0, sacc);
    #pragma unroll
    for (int v = 0; v < 8; ++v) {
      const float p = __expf(sacc[v] - rm[v]) * ri[v];
      softb[(wave * 16 + v + 8 * half) * SF_LD + t * 16 + r] = f2bfu(p);
    }
    wait_async0();
    __syncthreads();
  }
  __syncthreads();   // xn/bb region now dead; phase 2 may overwrite it

  // ---- phase 2: out = soft @ new_m, 4 channel groups of 128 ----
  const unsigned short* srowB = softb + (wave * 16 + r) * SF_LD;
  float* obase = out + (size_t)b * (512u * 4096u) + hw;
  for (int cg = 0; cg < 4; ++cg) {
    const int c0 = cg * 128;
    f32x8 acc[8];
    f32x8 z = {};
    #pragma unroll
    for (int ct = 0; ct < 8; ++ct) acc[ct] = z;

    stage_panel(nmT, c0, 0, bl0);
    wait_async0();
    __syncthreads();
    for (int kb = 0; kb < 8; ++kb) {
      const int k0 = kb * 128;
      const unsigned short* cur = (kb & 1) ? bl1 : bl0;
      unsigned short*       nxt = (kb & 1) ? bl0 : bl1;
      if (kb < 7) stage_panel(nmT, c0, k0 + 128, nxt);   // async, overlaps
      bf16x16 af[4];
      #pragma unroll
      for (int kk = 0; kk < 4; ++kk) af[kk] = load_a_frag(srowB + k0 + kk * 32, half);
      // 32 pipelined WMMAs: i = ct*4 + kk
      bf16x16 bcur = load_b_frag(cur + r * BL_LD, half);
      #pragma unroll
      for (int i = 0; i < 31; ++i) {
        const int ct = i >> 2, kk = i & 3;
        const int j = i + 1, ct2 = j >> 2, kk2 = j & 3;
        bf16x16 bnxt = load_b_frag(cur + (ct2 * 16 + r) * BL_LD + kk2 * 32, half);
        acc[ct] = wmma_bf16(af[kk], bcur, acc[ct]);
        bcur = bnxt;
      }
      acc[7] = wmma_bf16(af[3], bcur, acc[7]);
      wait_async0();
      __syncthreads();
    }
    // store D tiles straight to NCHW (each 64B line = 16 w's of one channel,
    // fully covered by this WG; L2 merges the partial-line writes)
    #pragma unroll
    for (int ct = 0; ct < 8; ++ct) {
      const int ch = c0 + ct * 16 + r;
      #pragma unroll
      for (int v = 0; v < 8; ++v) {
        const int w = wave * 16 + v + 8 * half;
        obase[(size_t)ch * 4096 + w] = acc[ct][v];
      }
    }
    __syncthreads();   // bl buffers reused next cg
  }
}

// ---------------------------------------------------------------------------
extern "C" void kernel_launch(void* const* d_in, const int* in_sizes, int n_in,
                              void* d_out, int out_size, void* d_ws, size_t ws_size,
                              hipStream_t stream) {
  const float* x = (const float*)d_in[0];   // [8,512,64,64]
  const float* m = (const float*)d_in[1];   // [1024,512]
  float* out = (float*)d_out;

  // workspace carve-up (~4.4 MiB)
  float*          counts = (float*)d_ws;                           // 1024
  float*          esum   = counts + 1024;                          // 1024*512
  unsigned short* mnbf   = (unsigned short*)(esum + 1024 * 512);   // 1 MiB
  unsigned short* nmT    = mnbf + 1024 * 512;                      // 1 MiB
  float*          rowMax = (float*)(nmT + 1024 * 512);             // 32768
  float*          rowSum = rowMax + 32768;                         // 32768
  float*          rnx    = rowSum + 32768;                         // 32768

  const int lds_k2 = 64 * XN_LD * 2 + 2 * 16 * BB_LD * 2 + 64 * 4;          // 100,096
  const int lds_k4 = 64 * XN_LD * 2 + 2 * 16 * BB_LD * 2 + 64 * SF_LD * 2;  // 232,960
  (void)hipFuncSetAttribute((const void*)SML_k_score,
                            hipFuncAttributeMaxDynamicSharedMemorySize, lds_k2);
  (void)hipFuncSetAttribute((const void*)SML_k_out,
                            hipFuncAttributeMaxDynamicSharedMemorySize, lds_k4);

  SML_k_zero  <<<2048, 256, 0, stream>>>(esum, counts);
  SML_k_normx <<< 512, 256, 0, stream>>>(x, rnx);
  SML_k_normm <<< 128, 256, 0, stream>>>(m, mnbf);
  SML_k_score <<< 512, 128, lds_k2, stream>>>(x, mnbf, rnx, rowMax, rowSum,
                                              counts, esum);
  SML_k_update<<<2048, 256, 0, stream>>>(m, esum, counts, nmT);
  SML_k_out   <<< 512, 128, lds_k4, stream>>>(x, mnbf, nmT, rnx, rowMax,
                                              rowSum, out);
}
